// BoundRayPatchEmbed_53824530153716
// MI455X (gfx1250) — compile-verified, matrix operand order checked by reading, not run on hardware
//
#include <hip/hip_runtime.h>
#include <hip/hip_bf16.h>

// Problem dims (match reference setup_inputs)
static constexpr int B = 16;
static constexpr int P = 512;
static constexpr int R = 8192;
static constexpr int D = 768;
static constexpr int M_TOTAL = B * R;          // 131072 flattened rays
static constexpr float NEG_INIT = -3.402823466e+38f;  // finfo(f32).min

typedef __attribute__((ext_vector_type(16))) _Float16     v16h;
typedef __attribute__((ext_vector_type(8)))  float        v8f;
typedef __attribute__((ext_vector_type(8)))  unsigned int v8u;

__device__ __forceinline__ float gelu_exact(float x) {
    return 0.5f * x * (1.0f + erff(x * 0.70710678118654752440f));
}

__device__ __forceinline__ unsigned int pack_h2(float a, float b) {
    _Float16 ha = (_Float16)a, hb = (_Float16)b;
    unsigned int lo = (unsigned int)__builtin_bit_cast(unsigned short, ha);
    unsigned int hi = (unsigned int)__builtin_bit_cast(unsigned short, hb);
    return lo | (hi << 16);
}

// Deterministic float atomic max via signed/unsigned int ordering trick.
__device__ __forceinline__ void atomicMaxF(float* addr, float val) {
    if (val >= 0.0f) {
        atomicMax((int*)addr, __float_as_int(val));
    } else {
        atomicMin((unsigned int*)addr, __float_as_uint(val));
    }
}

// CDNA5 async global->LDS copy (16B per lane), tracked by ASYNCcnt.
__device__ __forceinline__ void async_copy_b128(unsigned lds_addr, const void* gptr) {
    asm volatile("global_load_async_to_lds_b128 %0, %1, off"
                 :: "v"(lds_addr), "v"(gptr) : "memory");
}
__device__ __forceinline__ void wait_asynccnt0() {
    asm volatile("s_wait_asynccnt 0x0" ::: "memory");
}

// ---------------------------------------------------------------------------
// Kernel 0: init outputs to -FLT_MAX and zero the per-patch ray counters.
// ---------------------------------------------------------------------------
__global__ void brpe_init_kernel(float* __restrict__ out, size_t total,
                                 int* __restrict__ counts, int ncounts) {
    size_t stride = (size_t)gridDim.x * blockDim.x;
    for (size_t i = (size_t)blockIdx.x * blockDim.x + threadIdx.x; i < total; i += stride)
        out[i] = NEG_INIT;
    for (size_t i = (size_t)blockIdx.x * blockDim.x + threadIdx.x; i < (size_t)ncounts; i += stride)
        counts[i] = 0;
}

// ---------------------------------------------------------------------------
// Kernel 0b: convert+pack W2 (f32, DxD) into half2 K-pairs:
// W2p[kp2 * D + n] = { f16(W2[2*kp2][n]), f16(W2[2*kp2+1][n]) }
// ---------------------------------------------------------------------------
__global__ void brpe_packW2_kernel(const float* __restrict__ W2,
                                   unsigned int* __restrict__ W2p) {
    size_t total = (size_t)(D / 2) * D;
    size_t stride = (size_t)gridDim.x * blockDim.x;
    for (size_t i = (size_t)blockIdx.x * blockDim.x + threadIdx.x; i < total; i += stride) {
        size_t kp2 = i / D, n = i - kp2 * D;
        W2p[i] = pack_h2(W2[(2 * kp2) * D + n], W2[(2 * kp2 + 1) * D + n]);
    }
}

// ---------------------------------------------------------------------------
// Kernel 1: per-batch proxy sphere center + radius (one block per batch).
// ---------------------------------------------------------------------------
__global__ void brpe_center_kernel(const float* __restrict__ centers,
                                   float* __restrict__ pc, float* __restrict__ pr) {
    const int b = blockIdx.x;
    const int tid = threadIdx.x;
    const float* c = centers + (size_t)b * P * 3;
    __shared__ float sx[256], sy[256], sz[256];
    __shared__ float cen[3];

    float x = 0.f, y = 0.f, z = 0.f;
    for (int p = tid; p < P; p += 256) {
        x += c[3 * p + 0]; y += c[3 * p + 1]; z += c[3 * p + 2];
    }
    sx[tid] = x; sy[tid] = y; sz[tid] = z;
    __syncthreads();
    for (int s = 128; s > 0; s >>= 1) {
        if (tid < s) { sx[tid] += sx[tid + s]; sy[tid] += sy[tid + s]; sz[tid] += sz[tid + s]; }
        __syncthreads();
    }
    if (tid == 0) {
        cen[0] = sx[0] / (float)P; cen[1] = sy[0] / (float)P; cen[2] = sz[0] / (float)P;
        pc[3 * b + 0] = cen[0]; pc[3 * b + 1] = cen[1]; pc[3 * b + 2] = cen[2];
    }
    __syncthreads();

    float mx = 0.f;
    for (int p = tid; p < P; p += 256) {
        float dx = c[3 * p + 0] - cen[0];
        float dy = c[3 * p + 1] - cen[1];
        float dz = c[3 * p + 2] - cen[2];
        mx = fmaxf(mx, sqrtf(dx * dx + dy * dy + dz * dz));
    }
    sx[tid] = mx;
    __syncthreads();
    for (int s = 128; s > 0; s >>= 1) {
        if (tid < s) sx[tid] = fmaxf(sx[tid], sx[tid + s]);
        __syncthreads();
    }
    if (tid == 0) pr[b] = fmaxf(sx[0], 0.001f) * 1.05f;
}

// ---------------------------------------------------------------------------
// Kernel 2: sphere anchor, nearest-center argmin, feature build.
// grid = (R/256, B)
// ---------------------------------------------------------------------------
__global__ void brpe_anchor_kernel(const float* __restrict__ centers,
                                   const float* __restrict__ ray_o,
                                   const float* __restrict__ ray_d,
                                   const float* __restrict__ ray_t,
                                   const float* __restrict__ ray_hit,
                                   const float* __restrict__ ray_n,
                                   const float* __restrict__ pc,
                                   const float* __restrict__ pr,
                                   float* __restrict__ feat_q,
                                   float* __restrict__ feat_a,
                                   int* __restrict__ assign,
                                   int* __restrict__ counts) {
    const int b = blockIdx.y;
    const int tid = threadIdx.x;
    __shared__ float cx[P], cy[P], cz[P];
    const float* c = centers + (size_t)b * P * 3;
    for (int p = tid; p < P; p += 256) {
        cx[p] = c[3 * p + 0]; cy[p] = c[3 * p + 1]; cz[p] = c[3 * p + 2];
    }
    __syncthreads();

    const int r = blockIdx.x * 256 + tid;
    const size_t g = (size_t)b * R + r;
    const float rox = ray_o[3 * g + 0], roy = ray_o[3 * g + 1], roz = ray_o[3 * g + 2];
    const float rdx = ray_d[3 * g + 0], rdy = ray_d[3 * g + 1], rdz = ray_d[3 * g + 2];
    const float cenx = pc[3 * b + 0], ceny = pc[3 * b + 1], cenz = pc[3 * b + 2];
    const float rad = pr[b];

    const float ocx = rox - cenx, ocy = roy - ceny, ocz = roz - cenz;
    const float EPS = 1e-6f;
    const float aa = rdx * rdx + rdy * rdy + rdz * rdz;
    const float bb = 2.0f * (ocx * rdx + ocy * rdy + ocz * rdz);
    const float cc = ocx * ocx + ocy * ocy + ocz * ocz - rad * rad;
    const float disc = bb * bb - 4.0f * aa * cc;
    const bool has_real = disc > 0.0f;
    const float sq = sqrtf(fmaxf(disc, 0.0f));
    const float denom = 2.0f * aa + EPS;
    const float t1 = (-bb - sq) / denom;
    const float t2 = (-bb + sq) / denom;
    const float t_pos = (t1 > EPS) ? t1 : ((t2 > EPS) ? t2 : 0.0f);
    const float t_closest = fmaxf(-bb / denom, 0.0f);
    const float t = ((t_pos > 0.0f) && has_real) ? t_pos : t_closest;
    const float ax = rox + t * rdx, ay = roy + t * rdy, az = roz + t * rdz;

    // argmin over 512 centers (first-min tiebreak like jnp.argmin)
    float best = 3.402823466e+38f;
    int bi = 0;
    for (int p = 0; p < P; ++p) {
        float dx = ax - cx[p], dy = ay - cy[p], dz = az - cz[p];
        float d2 = dx * dx + dy * dy + dz * dz;
        if (d2 < best) { best = d2; bi = p; }
    }
    const float relx = ax - cx[bi], rely = ay - cy[bi], relz = az - cz[bi];

    float* fq = feat_q + g * 6;
    fq[0] = relx; fq[1] = rely; fq[2] = relz; fq[3] = rdx; fq[4] = rdy; fq[5] = rdz;
    float* fa = feat_a + g * 11;
    fa[0] = relx; fa[1] = rely; fa[2] = relz; fa[3] = rdx; fa[4] = rdy; fa[5] = rdz;
    fa[6] = ray_hit[g]; fa[7] = ray_t[g];
    fa[8] = ray_n[3 * g + 0]; fa[9] = ray_n[3 * g + 1]; fa[10] = ray_n[3 * g + 2];
    assign[g] = bi;
    atomicAdd(&counts[b * P + bi], 1);
}

// ---------------------------------------------------------------------------
// Kernel 3: fused MLP (layer1 VALU + layer2 WMMA) + scatter-max epilogue.
// Block: 256 threads (8 wave32). Block tile: 128(M) x 128(N). K step = 32.
// Each wave: 4 M-subtiles x 2 N-subtiles = 8 WMMAs per k-step (64 acc VGPRs).
// B tile staged via GLOBAL_LOAD_ASYNC_TO_LDS_B128 (ASYNCcnt), overlapped with
// the on-the-fly layer-1 compute of the A tile.
// grid = (M_TOTAL/128, D/128)
// ---------------------------------------------------------------------------
template <int IN>
__global__ void brpe_mlp_scatter_kernel(const float* __restrict__ feat,
                                        const float* __restrict__ W1,
                                        const float* __restrict__ b1,
                                        const unsigned int* __restrict__ W2p,
                                        const float* __restrict__ b2,
                                        const int* __restrict__ assign,
                                        float* __restrict__ out) {
    constexpr int IN_PAD = (IN <= 8) ? 8 : 12;
    const int tid = threadIdx.x;
    const int mBase = blockIdx.x * 128;
    const int nBase = blockIdx.y * 128;

    __shared__ float featS[128][IN_PAD];
    __shared__ float w1All[IN * D];                      // full layer-1 weights (<=34KB)
    __shared__ float b1All[D];
    __shared__ __align__(16) unsigned int aP[128][17];   // 128x32 f16 A tile as k-pairs (+pad)
    __shared__ __align__(16) unsigned int bP[16][132];   // 32x128 f16 B tile as k-pairs (row stride 528B)

    // One-time staging: features for this M-tile, full W1/b1.
    for (int i = tid; i < 128 * IN; i += 256) {
        int row = i / IN, j = i - row * IN;
        featS[row][j] = feat[(size_t)(mBase + row) * IN + j];
    }
    for (int i = tid; i < IN * D; i += 256) w1All[i] = W1[i];
    for (int i = tid; i < D; i += 256) b1All[i] = b1[i];

    // Per-thread async-copy coordinates for the B tile (2 x 16B chunks each).
    const int kk2a = tid >> 5,        nn4a = (tid & 31) * 4;        // rows 0..7
    const int kk2b = (tid >> 5) + 8,  nn4b = nn4a;                  // rows 8..15
    const unsigned ldsA = (unsigned)(size_t)&bP[kk2a][nn4a];
    const unsigned ldsB = (unsigned)(size_t)&bP[kk2b][nn4b];

    const int lane   = tid & 31;
    const int wave   = tid >> 5;
    const int nGroup = wave & 3;          // 4 N-groups of 32 -> 128 cols
    const int mWave  = (wave >> 2) * 64;  // 2 M-groups of 64 -> 128 rows
    const int lcol   = lane & 15;
    const int lhalf  = lane >> 4;

    v8f acc[4][2];
#pragma unroll
    for (int t = 0; t < 4; ++t) {
        acc[t][0] = v8f{};
        acc[t][1] = v8f{};
    }

    for (int k0 = 0; k0 < D; k0 += 32) {
        const int k0h = k0 >> 1;
        __syncthreads();  // previous iteration's aP/bP readers are done
        // Fire-and-forget async DMA of the packed B tile into LDS.
        async_copy_b128(ldsA, W2p + (size_t)(k0h + kk2a) * D + nBase + nn4a);
        async_copy_b128(ldsB, W2p + (size_t)(k0h + kk2b) * D + nBase + nn4b);
        // Layer-1 on the fly (overlaps the async copies): two adjacent k per
        // thread-step, packed f16 store into the A tile.
        for (int i2 = tid; i2 < 128 * 16; i2 += 256) {
            int row = i2 >> 4, kk2 = i2 & 15;
            int k = k0 + 2 * kk2;
            float s0 = b1All[k], s1 = b1All[k + 1];
#pragma unroll
            for (int j = 0; j < IN; ++j) {
                float f = featS[row][j];
                s0 += f * w1All[j * D + k];
                s1 += f * w1All[j * D + k + 1];
            }
            aP[row][kk2] = pack_h2(gelu_exact(s0), gelu_exact(s1));
        }
        wait_asynccnt0();   // B tile resident in LDS
        __syncthreads();

        // Fragment build (packed dword loads only) + 8 WMMAs per wave.
        v8u bu0, bu1;
#pragma unroll
        for (int v = 0; v < 8; ++v) {
            int kp2 = ((v < 4) ? v : v + 4) + lhalf * 4;
            bu0[v] = bP[kp2][nGroup * 32 + lcol];
            bu1[v] = bP[kp2][nGroup * 32 + 16 + lcol];
        }
        const v16h bf0 = __builtin_bit_cast(v16h, bu0);
        const v16h bf1 = __builtin_bit_cast(v16h, bu1);
#pragma unroll
        for (int t = 0; t < 4; ++t) {
            int row = mWave + t * 16 + lcol;
            v8u au;
#pragma unroll
            for (int v = 0; v < 8; ++v) {
                int kp2 = ((v < 4) ? v : v + 4) + lhalf * 4;
                au[v] = aP[row][kp2];
            }
            const v16h af = __builtin_bit_cast(v16h, au);
            acc[t][0] = __builtin_amdgcn_wmma_f32_16x16x32_f16(
                false, af, false, bf0, (short)0, acc[t][0], false, false);
            acc[t][1] = __builtin_amdgcn_wmma_f32_16x16x32_f16(
                false, af, false, bf1, (short)0, acc[t][1], false, false);
        }
    }

    // Epilogue: bias add + scatter-max into out[b, assign[m], :].
#pragma unroll
    for (int u = 0; u < 2; ++u) {
        const int n = nBase + nGroup * 32 + u * 16 + lcol;
        const float bias = b2[n];
#pragma unroll
        for (int t = 0; t < 4; ++t) {
#pragma unroll
            for (int r = 0; r < 8; ++r) {
                int mg = mBase + mWave + t * 16 + r + lhalf * 8;  // C layout: M = r + 8*(lane>=16)
                int bIdx = mg >> 13;                               // mg / R (R = 8192)
                int p = assign[mg];
                float v = acc[t][u][r] + bias;
                atomicMaxF(&out[((size_t)bIdx * P + p) * D + n], v);
            }
        }
    }
}

// ---------------------------------------------------------------------------
// Kernel 4: mask empty patches, emit has_ray. One block per (b,p).
// ---------------------------------------------------------------------------
__global__ void brpe_finalize_kernel(const int* __restrict__ counts,
                                     float* __restrict__ out_q,
                                     float* __restrict__ out_a,
                                     float* __restrict__ has_ray) {
    const int bp = blockIdx.x;
    const int cnt = counts[bp];
    if (threadIdx.x == 0) has_ray[bp] = (cnt > 0) ? 1.0f : 0.0f;
    if (cnt > 0) return;  // values already valid (maxed); mask==1 keeps them
    float* oq = out_q + (size_t)bp * D;
    float* oa = out_a + (size_t)bp * D;
    for (int d = threadIdx.x; d < D; d += blockDim.x) { oq[d] = 0.0f; oa[d] = 0.0f; }
}

// ---------------------------------------------------------------------------
extern "C" void kernel_launch(void* const* d_in, const int* in_sizes, int n_in,
                              void* d_out, int out_size, void* d_ws, size_t ws_size,
                              hipStream_t stream) {
    const float* centers = (const float*)d_in[0];
    const float* ray_o   = (const float*)d_in[1];
    const float* ray_d   = (const float*)d_in[2];
    const float* ray_t   = (const float*)d_in[3];
    const float* ray_hit = (const float*)d_in[4];
    const float* ray_n   = (const float*)d_in[5];
    const float* Wq1 = (const float*)d_in[6];
    const float* bq1 = (const float*)d_in[7];
    const float* Wq2 = (const float*)d_in[8];
    const float* bq2 = (const float*)d_in[9];
    const float* Wa1 = (const float*)d_in[10];
    const float* ba1 = (const float*)d_in[11];
    const float* Wa2 = (const float*)d_in[12];
    const float* ba2 = (const float*)d_in[13];

    float* out_q   = (float*)d_out;
    float* out_a   = out_q + (size_t)B * P * D;
    float* has_ray = out_a + (size_t)B * P * D;

    // Workspace carve-up (256B aligned blocks)
    char* ws = (char*)d_ws;
    float* pc     = (float*)(ws);                       // B*3
    float* pr     = (float*)(ws + 256);                 // B
    int*   assign = (int*)  (ws + 512);                 // B*R
    int*   counts = (int*)  (ws + 512 + (size_t)B * R * 4);                 // B*P
    float* feat_q = (float*)(ws + 512 + (size_t)B * R * 4 + (size_t)B * P * 4);
    float* feat_a = feat_q + (size_t)B * R * 6;
    unsigned int* W2p_q = (unsigned int*)(feat_a + (size_t)B * R * 11);     // (D/2)*D
    unsigned int* W2p_a = W2p_q + (size_t)(D / 2) * D;
    (void)in_sizes; (void)n_in; (void)out_size; (void)ws_size;

    brpe_init_kernel<<<2048, 256, 0, stream>>>(out_q, (size_t)2 * B * P * D, counts, B * P);
    brpe_packW2_kernel<<<1152, 256, 0, stream>>>(Wq2, W2p_q);
    brpe_packW2_kernel<<<1152, 256, 0, stream>>>(Wa2, W2p_a);
    brpe_center_kernel<<<B, 256, 0, stream>>>(centers, pc, pr);
    brpe_anchor_kernel<<<dim3(R / 256, B), 256, 0, stream>>>(
        centers, ray_o, ray_d, ray_t, ray_hit, ray_n, pc, pr,
        feat_q, feat_a, assign, counts);
    brpe_mlp_scatter_kernel<6><<<dim3(M_TOTAL / 128, D / 128), 256, 0, stream>>>(
        feat_q, Wq1, bq1, W2p_q, bq2, assign, out_q);
    brpe_mlp_scatter_kernel<11><<<dim3(M_TOTAL / 128, D / 128), 256, 0, stream>>>(
        feat_a, Wa1, ba1, W2p_a, ba2, assign, out_a);
    brpe_finalize_kernel<<<B * P, 256, 0, stream>>>(counts, out_q, out_a, has_ray);
}
